// SpikingDecoder_36764920053774
// MI455X (gfx1250) — compile-verified, matrix operand order checked by reading, not run on hardware
//
#include <hip/hip_runtime.h>
#include <hip/hip_bf16.h>
#include <math.h>

#define L_  2
#define B_  16
#define S_  32
#define SE_ 128
#define H_  512
#define E_  512
#define IN_ 1024   // E + H
#define G4_ 2048   // 4*H

typedef __bf16 bf16;
typedef __attribute__((ext_vector_type(16))) __bf16 v16bf;
typedef __attribute__((ext_vector_type(8)))  __bf16 v8bf;
typedef __attribute__((ext_vector_type(8)))  float  v8f;

// ---------------- WMMA helpers (CDNA5 16-bit fragment layout) ----------------
// A 16x32 (MxK): lane half=lane>>4, row=lane&15; 16 elems = two contiguous
// 8-elem chunks at k0+half*8 and k0+16+half*8.  B 32x16 (KxN) mirrors with
// N=lane&15 when weights are stored row-major as W[N][K] (computing X @ W^T).
__device__ __forceinline__ v16bf ldfrag(const bf16* __restrict__ p) {
  v8bf lo = *(const v8bf*)(p);
  v8bf hi = *(const v8bf*)(p + 16);
  return __builtin_shufflevector(lo, hi, 0,1,2,3,4,5,6,7,8,9,10,11,12,13,14,15);
}

__device__ __forceinline__ void gemm_acc(v8f& acc,
    const bf16* __restrict__ A, int lda, int m0,
    const bf16* __restrict__ W, int ldw, int n0, int K) {
  const int lane = threadIdx.x & 31;
  const int half = lane >> 4;
  const int r    = lane & 15;
  const bf16* ap = A + (m0 + r) * lda + half * 8;
  const bf16* wp = W + (n0 + r) * ldw + half * 8;
  for (int k = 0; k < K; k += 32) {
    v16bf a = ldfrag(ap + k);
    v16bf b = ldfrag(wp + k);
    acc = __builtin_amdgcn_wmma_f32_16x16x32_bf16(false, a, false, b,
                                                  (short)0, acc, false, false);
  }
}

__device__ __forceinline__ float sigf(float x) { return 1.0f / (1.0f + expf(-x)); }

// ---------------- setup kernels ----------------
__global__ void k_cvt_bf16(const float* __restrict__ s, bf16* __restrict__ d, int n) {
  for (int i = blockIdx.x * blockDim.x + threadIdx.x; i < n; i += gridDim.x * blockDim.x)
    d[i] = (bf16)s[i];
}

__global__ void k_embed(const int* __restrict__ tgt, const float* __restrict__ emb,
                        const float* __restrict__ al, const float* __restrict__ ga,
                        const float* __restrict__ be, bf16* __restrict__ xb) {
  int i = blockIdx.x * blockDim.x + threadIdx.x;
  if (i >= B_ * S_ * E_) return;
  int e = i & (E_ - 1);
  int bs = i >> 9;
  float v = emb[(long)tgt[bs] * E_ + e];
  xb[i] = (bf16)(ga[e] * tanhf(al[0] * v) + be[e]);
}

__global__ void k_init(const float* __restrict__ esyn, const float* __restrict__ emem,
                       float* __restrict__ syn, float* __restrict__ mem,
                       bf16* __restrict__ memb) {
  int i = blockIdx.x * blockDim.x + threadIdx.x;
  if (i >= L_ * B_ * H_) return;
  syn[i] = esyn[i];
  float m = emem[i];
  mem[i] = m;
  memb[i] = (bf16)m;
}

// pre_enc[b*SE+s][h] = enc @ We^T + attn_b   (time invariant)
__global__ void k_encproj(const bf16* __restrict__ encb, const bf16* __restrict__ We,
                          const float* __restrict__ abias, float* __restrict__ pre) {
  int m0 = blockIdx.x * 16;          // row tile over B*SE = 2048
  int n0 = blockIdx.y * 16;          // col tile over H = 512
  v8f acc = {};
  gemm_acc(acc, encb, H_, m0, We, 2 * H_, n0, H_);
  int lane = threadIdx.x & 31;
  int n = n0 + (lane & 15);
  int mb = (lane >> 4) * 8;
  float bn = abias[n];
  #pragma unroll
  for (int v = 0; v < 8; ++v)
    pre[(long)(m0 + mb + v) * H_ + n] = acc[v] + bn;
}

// ---------------- per-step kernels ----------------
// hproj = mem @ Wh^T  (16 x 512, K=512)
__global__ void k_hproj(const bf16* __restrict__ memb, const bf16* __restrict__ Wh,
                        float* __restrict__ hproj) {
  int n0 = blockIdx.x * 16;
  v8f acc = {};
  gemm_acc(acc, memb, H_, 0, Wh, 2 * H_, n0, H_);
  int lane = threadIdx.x & 31;
  int n = n0 + (lane & 15);
  int mb = (lane >> 4) * 8;
  #pragma unroll
  for (int v = 0; v < 8; ++v)
    hproj[(mb + v) * H_ + n] = acc[v];
}

// score[b][s] = sum_h tanh(hproj[b][h] + pre[b,s,h]) * vW[h] + vb   (one wave/row)
__global__ void k_score(const float* __restrict__ hproj, const float* __restrict__ pre,
                        const float* __restrict__ vW, const float* __restrict__ vb,
                        float* __restrict__ score) {
  int wave = threadIdx.x >> 5;
  int lane = threadIdx.x & 31;
  int row = blockIdx.x * 8 + wave;             // row in [0, B*SE)
  int b = row >> 7;
  const float* hp = hproj + b * H_;
  const float* pp = pre + (long)row * H_;
  float acc = 0.f;
  for (int h = lane; h < H_; h += 32)
    acc += tanhf(hp[h] + pp[h]) * vW[h];
  #pragma unroll
  for (int o = 16; o > 0; o >>= 1) acc += __shfl_xor(acc, o, 32);
  if (lane == 0) score[row] = acc + vb[0];
}

// softmax over SE per batch; ctx = w @ enc; DyT; pack xin = [inp | ctx] (bf16)
// and (layer 0) fcin[:,512:1536] = xin
__global__ void k_ctx(const float* __restrict__ score, const float* __restrict__ encf,
                      const float* __restrict__ aAl, const float* __restrict__ aGa,
                      const float* __restrict__ aBe,
                      const bf16* __restrict__ inp, int ldinp,
                      bf16* __restrict__ xin, bf16* __restrict__ fcin) {
  __shared__ float sc[SE_];
  __shared__ float red[SE_];
  int b = blockIdx.x, tid = threadIdx.x;
  float s = score[b * SE_ + tid];
  sc[tid] = s; red[tid] = s;
  __syncthreads();
  for (int o = 64; o > 0; o >>= 1) { if (tid < o) red[tid] = fmaxf(red[tid], red[tid + o]); __syncthreads(); }
  float mx = red[0];
  __syncthreads();
  float ev = expf(s - mx);
  sc[tid] = ev; red[tid] = ev;
  __syncthreads();
  for (int o = 64; o > 0; o >>= 1) { if (tid < o) red[tid] += red[tid + o]; __syncthreads(); }
  float inv = 1.0f / red[0];
  __syncthreads();
  sc[tid] = ev * inv;                // softmax weights
  __syncthreads();

  float alpha = aAl[0];
  const float* eb = encf + (long)b * SE_ * H_;
  #pragma unroll
  for (int j = 0; j < 4; ++j) {
    int c = tid + j * SE_;           // 0..511
    // copy input half of xin
    bf16 iv = inp[(long)b * ldinp + c];
    xin[b * IN_ + c] = iv;
    if (fcin) fcin[b * (H_ + IN_) + H_ + c] = iv;
    // context half
    float acc = 0.f;
    for (int ss = 0; ss < SE_; ++ss)
      acc += sc[ss] * eb[(long)ss * H_ + c];
    float d = aGa[c] * tanhf(alpha * acc) + aBe[c];
    xin[b * IN_ + H_ + c] = (bf16)d;
    if (fcin) fcin[b * (H_ + IN_) + H_ + H_ + c] = (bf16)d;
  }
}

// gates = xin @ W_ih^T + mem @ W_hh^T + b_ih + b_hh   (16 x 2048)
__global__ void k_gates(const bf16* __restrict__ xin, const bf16* __restrict__ Wih,
                        const bf16* __restrict__ memb, const bf16* __restrict__ Whh,
                        const float* __restrict__ bih, const float* __restrict__ bhh,
                        float* __restrict__ gates) {
  int n0 = blockIdx.x * 16;
  v8f acc = {};
  gemm_acc(acc, xin,  IN_, 0, Wih, IN_, n0, IN_);
  gemm_acc(acc, memb, H_,  0, Whh, H_,  n0, H_);
  int lane = threadIdx.x & 31;
  int n = n0 + (lane & 15);
  int mb = (lane >> 4) * 8;
  float bn = bih[n] + bhh[n];
  #pragma unroll
  for (int v = 0; v < 8; ++v)
    gates[(mb + v) * G4_ + n] = acc[v] + bn;
}

// LSTM cell + heaviside spike; updates syn/mem state; layer 1 writes d_out
__global__ void k_cell(const float* __restrict__ gates, float* __restrict__ syn,
                       float* __restrict__ mem, bf16* __restrict__ memb,
                       bf16* __restrict__ spkb, const float* __restrict__ thrp, int l,
                       float* __restrict__ outp, int t) {
  int i = blockIdx.x * blockDim.x + threadIdx.x;
  if (i >= B_ * H_) return;
  int b = i >> 9, h = i & (H_ - 1);
  const float* g = gates + b * G4_;
  float ig = sigf(g[h]);
  float fg = sigf(g[H_ + h]);
  float gg = tanhf(g[2 * H_ + h]);
  float og = sigf(g[3 * H_ + h]);
  float c  = fg * syn[i] + ig * gg;
  float hN = og * tanhf(c);
  float spk = (hN - thrp[l] > 0.f) ? 1.0f : 0.0f;
  syn[i] = c;
  mem[i] = hN;
  memb[i] = (bf16)hN;
  spkb[i] = (bf16)spk;
  if (outp) outp[(long)b * S_ * H_ + (long)t * H_ + h] = spk;
}

// fcin[:,0:512] = spk @ lin^T + lin_b
__global__ void k_fc1(const bf16* __restrict__ spkb, const bf16* __restrict__ linW,
                      const float* __restrict__ linb, bf16* __restrict__ fcin) {
  int n0 = blockIdx.x * 16;
  v8f acc = {};
  gemm_acc(acc, spkb, H_, 0, linW, H_, n0, H_);
  int lane = threadIdx.x & 31;
  int n = n0 + (lane & 15);
  int mb = (lane >> 4) * 8;
  float bn = linb[n];
  #pragma unroll
  for (int v = 0; v < 8; ++v)
    fcin[(mb + v) * (H_ + IN_) + n] = (bf16)(acc[v] + bn);
}

// layer output = DyT(fcin @ mixer^T + mixer_b)
__global__ void k_fc2(const bf16* __restrict__ fcin, const bf16* __restrict__ mixW,
                      const float* __restrict__ mixb, const float* __restrict__ fAl,
                      const float* __restrict__ fGa, const float* __restrict__ fBe,
                      bf16* __restrict__ lout) {
  int n0 = blockIdx.x * 16;
  v8f acc = {};
  gemm_acc(acc, fcin, H_ + IN_, 0, mixW, H_ + IN_, n0, H_ + IN_);
  int lane = threadIdx.x & 31;
  int n = n0 + (lane & 15);
  int mb = (lane >> 4) * 8;
  float bn = mixb[n], al = fAl[0], ga = fGa[n], be = fBe[n];
  #pragma unroll
  for (int v = 0; v < 8; ++v)
    lout[(mb + v) * H_ + n] = (bf16)(ga * tanhf(al * (acc[v] + bn)) + be);
}

// ---------------- host launch ----------------
static inline size_t alignup(size_t x) { return (x + 255) & ~(size_t)255; }

extern "C" void kernel_launch(void* const* d_in, const int* in_sizes, int n_in,
                              void* d_out, int out_size, void* d_ws, size_t ws_size,
                              hipStream_t stream) {
  const int*   tgt    = (const int*)  d_in[0];
  const float* enc    = (const float*)d_in[1];
  const float* esyn   = (const float*)d_in[2];
  const float* emem   = (const float*)d_in[3];
  const float* emb    = (const float*)d_in[4];
  const float* eAl    = (const float*)d_in[5];
  const float* eGa    = (const float*)d_in[6];
  const float* eBe    = (const float*)d_in[7];
  const float* attnW  = (const float*)d_in[8];
  const float* attnB  = (const float*)d_in[9];
  const float* vW     = (const float*)d_in[10];
  const float* vb     = (const float*)d_in[11];
  const float* aAl    = (const float*)d_in[12];
  const float* aGa    = (const float*)d_in[13];
  const float* aBe    = (const float*)d_in[14];
  const float* Wih    = (const float*)d_in[15];
  const float* Whh    = (const float*)d_in[16];
  const float* bih    = (const float*)d_in[17];
  const float* bhh    = (const float*)d_in[18];
  const float* thr    = (const float*)d_in[19];
  const float* linW   = (const float*)d_in[21];
  const float* linb   = (const float*)d_in[22];
  const float* mixW   = (const float*)d_in[23];
  const float* mixb   = (const float*)d_in[24];
  const float* fAl    = (const float*)d_in[25];
  const float* fGa    = (const float*)d_in[26];
  const float* fBe    = (const float*)d_in[27];
  float* outp = (float*)d_out;

  // workspace layout
  char* w = (char*)d_ws;
  size_t off = 0;
  auto take = [&](size_t bytes) { void* p = w + off; off = alignup(off + bytes); return p; };
  const int nEnc  = L_ * B_ * SE_ * H_;
  const int nAttW = L_ * H_ * 2 * H_;
  const int nWih  = L_ * G4_ * IN_;
  const int nWhh  = L_ * G4_ * H_;
  const int nLin  = H_ * H_;
  const int nMix  = H_ * (H_ + IN_);

  bf16*  xb      = (bf16*) take((size_t)B_ * S_ * E_ * sizeof(bf16));
  bf16*  encb    = (bf16*) take((size_t)nEnc  * sizeof(bf16));
  bf16*  attnWb  = (bf16*) take((size_t)nAttW * sizeof(bf16));
  bf16*  Wihb    = (bf16*) take((size_t)nWih  * sizeof(bf16));
  bf16*  Whhb    = (bf16*) take((size_t)nWhh  * sizeof(bf16));
  bf16*  linWb   = (bf16*) take((size_t)nLin  * sizeof(bf16));
  bf16*  mixWb   = (bf16*) take((size_t)nMix  * sizeof(bf16));
  float* pre     = (float*)take((size_t)nEnc  * sizeof(float));
  float* syn     = (float*)take((size_t)L_ * B_ * H_ * sizeof(float));
  float* mem     = (float*)take((size_t)L_ * B_ * H_ * sizeof(float));
  bf16*  memb    = (bf16*) take((size_t)L_ * B_ * H_ * sizeof(bf16));
  float* hproj   = (float*)take((size_t)B_ * H_ * sizeof(float));
  float* score   = (float*)take((size_t)B_ * SE_ * sizeof(float));
  bf16*  xinb    = (bf16*) take((size_t)B_ * IN_ * sizeof(bf16));
  float* gates   = (float*)take((size_t)B_ * G4_ * sizeof(float));
  bf16*  spkb    = (bf16*) take((size_t)B_ * H_ * sizeof(bf16));
  bf16*  fcinb   = (bf16*) take((size_t)B_ * (H_ + IN_) * sizeof(bf16));
  bf16*  loutb   = (bf16*) take((size_t)B_ * H_ * sizeof(bf16));
  (void)ws_size; (void)n_in; (void)in_sizes; (void)out_size;

  // ---- one-time (per launch) setup ----
  auto cvt = [&](const float* s, bf16* d, int n) {
    k_cvt_bf16<<<(n + 255) / 256, 256, 0, stream>>>(s, d, n);
  };
  cvt(attnW, attnWb, nAttW);
  cvt(Wih,   Wihb,   nWih);
  cvt(Whh,   Whhb,   nWhh);
  cvt(linW,  linWb,  nLin);
  cvt(mixW,  mixWb,  nMix);
  cvt(enc,   encb,   nEnc);

  k_embed<<<(B_ * S_ * E_ + 255) / 256, 256, 0, stream>>>(tgt, emb, eAl, eGa, eBe, xb);
  k_init<<<(L_ * B_ * H_ + 255) / 256, 256, 0, stream>>>(esyn, emem, syn, mem, memb);

  for (int l = 0; l < L_; ++l) {
    dim3 g((B_ * SE_) / 16, H_ / 16);
    k_encproj<<<g, 32, 0, stream>>>(encb + (size_t)l * B_ * SE_ * H_,
                                    attnWb + (size_t)l * H_ * 2 * H_ + H_,
                                    attnB + l * H_,
                                    pre + (size_t)l * B_ * SE_ * H_);
  }

  // ---- sequential scan over time ----
  for (int t = 0; t < S_; ++t) {
    for (int l = 0; l < L_; ++l) {
      const bf16* inp   = (l == 0) ? (xb + (size_t)t * E_) : loutb;
      const int   ldinp = (l == 0) ? (S_ * E_) : H_;
      bf16* memb_l = memb + (size_t)l * B_ * H_;
      float* syn_l = syn + (size_t)l * B_ * H_;
      float* mem_l = mem + (size_t)l * B_ * H_;

      k_hproj<<<H_ / 16, 32, 0, stream>>>(memb_l,
                                          attnWb + (size_t)l * H_ * 2 * H_, hproj);
      k_score<<<(B_ * SE_) / 8, 256, 0, stream>>>(hproj,
                                                  pre + (size_t)l * B_ * SE_ * H_,
                                                  vW + l * H_, vb + l, score);
      k_ctx<<<B_, SE_, 0, stream>>>(score, enc + (size_t)l * B_ * SE_ * H_,
                                    aAl + l, aGa + l * H_, aBe + l * H_,
                                    inp, ldinp, xinb,
                                    (l == 0) ? fcinb : (bf16*)nullptr);
      k_gates<<<G4_ / 16, 32, 0, stream>>>(xinb, Wihb + (size_t)l * G4_ * IN_,
                                           memb_l, Whhb + (size_t)l * G4_ * H_,
                                           bih + l * G4_, bhh + l * G4_, gates);
      k_cell<<<(B_ * H_ + 255) / 256, 256, 0, stream>>>(
          gates, syn_l, mem_l, memb_l, spkb, thr, l,
          (l == L_ - 1) ? outp : (float*)nullptr, t);
      if (l == 0) {
        k_fc1<<<H_ / 16, 32, 0, stream>>>(spkb, linWb, linb, fcinb);
        k_fc2<<<H_ / 16, 32, 0, stream>>>(fcinb, mixWb, mixb, fAl, fGa, fBe, loutb);
      }
    }
  }
}
